// Inter_50517405336004
// MI455X (gfx1250) — matrix operand order repaired and checked by reading, not compiled, not run on hardware
//
#include <hip/hip_runtime.h>
#include <hip/hip_bf16.h>

#define DEV __device__ __forceinline__

typedef __attribute__((ext_vector_type(16))) __bf16 v16bf;
typedef __attribute__((ext_vector_type(8)))  __bf16 v8bf;
typedef __attribute__((ext_vector_type(2)))  __bf16 v2bf;
typedef __attribute__((ext_vector_type(8)))  float  v8f;

static constexpr int Lseq = 1024, Dm = 1024, Hn = 8, Kh = 128, OUTN = 64;
static constexpr float QK_SCALE = 0.29730177875068026f; // 128^-0.25

DEV v8f zero8() {
  v8f r;
#pragma unroll
  for (int i = 0; i < 8; ++i) r[i] = 0.f;
  return r;
}

// Assemble a 16-half A/B fragment from two 16-byte chunks (per ISA A layout).
DEV v16bf frag2(const __bf16* lo, const __bf16* hi) {
  v8bf a = *(const v8bf*)lo;
  v8bf b = *(const v8bf*)hi;
  v16bf r;
#pragma unroll
  for (int i = 0; i < 8; ++i) { r[i] = a[i]; r[i + 8] = b[i]; }
  return r;
}

DEV v8f wmma_bf16(v16bf a, v16bf b, v8f c) {
  // (neg_a, A, neg_b, B, c_mod, C, reuse_a, reuse_b)
  return __builtin_amdgcn_wmma_f32_16x16x32_bf16(false, a, false, b, (short)0, c,
                                                 false, false);
}

// Async global->LDS copy (ASYNCcnt path, VGLOBAL op 98). LDS byte address is
// the low 32 bits of the generic pointer (ISA §10.2 LDS aperture truncation).
DEV void async_ld_lds_b128(__bf16* lds, const __bf16* gptr) {
  unsigned la = (unsigned)(size_t)lds;
  asm volatile("global_load_async_to_lds_b128 %0, %1, off"
               :: "v"(la), "v"(gptr) : "memory");
}
DEV void wait_asynccnt0() { asm volatile("s_wait_asynccnt 0x0" ::: "memory"); }

// ---------------------------------------------------------------- cvt weights
__global__ __launch_bounds__(256) void cvt_f32_bf16(const float* __restrict__ src,
                                                    __bf16* __restrict__ dst, int n) {
  int i = (blockIdx.x * 256 + threadIdx.x) * 4;
  if (i + 3 < n) {
    float4 v = *(const float4*)(src + i);
    dst[i] = (__bf16)v.x; dst[i + 1] = (__bf16)v.y;
    dst[i + 2] = (__bf16)v.z; dst[i + 3] = (__bf16)v.w;
  }
}

// ------------------------------------------------------------ C = A @ W^T GEMM
// A: [M, Kd] (f32 or bf16), W: [N, Kd] bf16 row-major (W rows are B columns).
// 128x128 block tile, 8 waves (4Mx2N), each wave 32x64 = 2x4 WMMA tiles, BK=32.
// MODE: 0 = f32 row-major out, 1 = bf16 row-major out,
//       2 = bf16 head-transposed out [b, h, kdim, l] (packed b128 stores).
template <typename TA, int MODE, bool HAS_BIAS>
__global__ __launch_bounds__(256) void gemm_xWt(const TA* __restrict__ A,
                                                const __bf16* __restrict__ W,
                                                const float* __restrict__ bias,
                                                float scale, void* __restrict__ Cptr,
                                                int Kd, int ldc) {
  __shared__ __bf16 As[128 * 32];  // fragment-permuted rows
  __shared__ __bf16 Bs[128 * 32];  // contiguous rows
  const int tid = threadIdx.x, lane = tid & 31, wave = tid >> 5;
  const int wm = (wave & 3) * 32, wn = (wave >> 2) * 64;
  const int lh = lane >> 4, lr = lane & 15;
  const long m0 = (long)blockIdx.x * 128, n0 = (long)blockIdx.y * 128;

  v8f acc[2][4];
#pragma unroll
  for (int a = 0; a < 2; ++a)
#pragma unroll
    for (int b = 0; b < 4; ++b) acc[a][b] = zero8();

  const int sr = tid >> 1, sh = (tid & 1) * 16;  // staging: row, k-half
  for (int k0 = 0; k0 < Kd; k0 += 32) {
    __syncthreads();
    {
      // --- stage A tile (convert to bf16, permute K into A-fragment order) ---
      float tmp[16];
      const TA* ap = A + (m0 + sr) * (long)Kd + k0 + sh;
      if constexpr (sizeof(TA) == 4) {
#pragma unroll
        for (int j = 0; j < 4; ++j) {
          float4 v = *(const float4*)((const float*)ap + j * 4);
          tmp[j * 4] = v.x; tmp[j * 4 + 1] = v.y;
          tmp[j * 4 + 2] = v.z; tmp[j * 4 + 3] = v.w;
        }
      } else {
#pragma unroll
        for (int j = 0; j < 2; ++j) {
          v8bf v = *(const v8bf*)((const __bf16*)ap + j * 8);
#pragma unroll
          for (int t = 0; t < 8; ++t) tmp[j * 8 + t] = (float)v[t];
        }
      }
      // adjacent even/odd k share a 32-bit slot in the permuted layout
#pragma unroll
      for (int j = 0; j < 16; j += 2) {
        int k = sh + j;  // fragment layout: {0-7,16-23} then {8-15,24-31}
        int pos = (k & 7) | (((k >> 4) & 1) << 3) | (((k >> 3) & 1) << 4);
        v2bf pk; pk[0] = (__bf16)tmp[j]; pk[1] = (__bf16)tmp[j + 1];
        *(v2bf*)&As[sr * 32 + pos] = pk;
      }
      // --- stage B tile (already bf16, contiguous K) ---
      const __bf16* wp = W + (n0 + sr) * (long)Kd + k0 + sh;
      *(v8bf*)&Bs[sr * 32 + sh] = *(const v8bf*)wp;
      *(v8bf*)&Bs[sr * 32 + sh + 8] = *(const v8bf*)(wp + 8);
      if (k0 + 32 < Kd) __builtin_prefetch(wp + 32, 0, 0);
    }
    __syncthreads();

    v16bf af[2], bf[4];
#pragma unroll
    for (int fm = 0; fm < 2; ++fm)
      af[fm] = *(const v16bf*)&As[(wm + fm * 16 + lr) * 32 + lh * 16];
#pragma unroll
    for (int fn = 0; fn < 4; ++fn)
      bf[fn] = frag2(&Bs[(wn + fn * 16 + lr) * 32 + lh * 8], &Bs[(wn + fn * 16 + lr) * 32 + 16 + lh * 8]);
#pragma unroll
    for (int fm = 0; fm < 2; ++fm)
#pragma unroll
      for (int fn = 0; fn < 4; ++fn)
        acc[fm][fn] = wmma_bf16(af[fm], bf[fn], acc[fm][fn]);
  }

#pragma unroll
  for (int fm = 0; fm < 2; ++fm)
#pragma unroll
    for (int fn = 0; fn < 4; ++fn) {
      long col = n0 + wn + fn * 16 + lr;
      if constexpr (MODE == 2) {
        // lane's 8 accum elements = 8 consecutive rows (consecutive l):
        // pack and store one b128 into [b, h, kdim, l].
        int h = (int)(col >> 7), kd = (int)(col & 127);
        long m = m0 + wm + fm * 16 + lh * 8;
        long bb = m >> 10, l = m & 1023;
        v8bf pk;
#pragma unroll
        for (int i = 0; i < 8; ++i) pk[i] = (__bf16)(acc[fm][fn][i] * scale);
        *(v8bf*)((__bf16*)Cptr +
                 ((bb * Hn + h) * (long)Kh + kd) * (long)Lseq + l) = pk;
      } else {
        float bv = 0.f;
        if constexpr (HAS_BIAS) bv = bias[col];
#pragma unroll
        for (int i = 0; i < 8; ++i) {
          long row = m0 + wm + fm * 16 + i + lh * 8;  // lanes>=16 -> M+8
          float v = acc[fm][fn][i] * scale + bv;
          if constexpr (MODE == 1)
            ((__bf16*)Cptr)[row * ldc + col] = (__bf16)v;
          else
            ((float*)Cptr)[row * ldc + col] = v;
        }
      }
    }
}

// --------------------------------------------------------- flash attention
// grid (L/128, H, B); 8 waves, each owns 16 query rows; 64-key blocks.
// K staged in LDS via async global->LDS copies; V read directly from the
// pre-transposed global layout [b, h, kdim, l] (contiguous B fragments).
__global__ __launch_bounds__(256) void attn_flash(const __bf16* __restrict__ Q,
                                                  const __bf16* __restrict__ Kt,
                                                  const __bf16* __restrict__ Vt,
                                                  __bf16* __restrict__ O) {
  __shared__ __bf16 Ks[64 * 128];      // keys row-major  [key][kdim]
  __shared__ __bf16 Ps[8 * 16 * 64];   // per-wave P relayout buffer
  const int tid = threadIdx.x, lane = tid & 31, wave = tid >> 5;
  const int lh = lane >> 4, lr = lane & 15;
  const int b = blockIdx.z, h = blockIdx.y;
  const long rowbase = (long)b * Lseq;
  const __bf16* vhead = Vt + ((long)b * Hn + h) * (long)Kh * Lseq;
  const int qr0 = blockIdx.x * 128 + wave * 16;

  // Q tile for this wave: 16 x 128, as 4 A-fragments (K chunks of 32)
  v16bf aq[4];
#pragma unroll
  for (int kc = 0; kc < 4; ++kc) {
    const __bf16* p = Q + (rowbase + qr0 + lr) * (long)Dm + h * Kh + kc * 32;
    aq[kc] = *(const v16bf*)(p + lh * 16);
  }

  v8f oacc[8];
#pragma unroll
  for (int i = 0; i < 8; ++i) oacc[i] = zero8();
  float mrow[8], lrow[8];
#pragma unroll
  for (int i = 0; i < 8; ++i) { mrow[i] = -3.0e38f; lrow[i] = 0.f; }

  const int srow = tid >> 2, sc = (tid & 3) * 32;
  for (int kb = 0; kb < Lseq / 64; ++kb) {
    __syncthreads();
    {  // async staging of the 64x128 key tile
      const __bf16* kp = Kt + (rowbase + kb * 64 + srow) * (long)Dm + h * Kh + sc;
      __bf16* kd = &Ks[srow * 128 + sc];
#pragma unroll
      for (int j = 0; j < 4; ++j) async_ld_lds_b128(kd + j * 8, kp + j * 8);
      wait_asynccnt0();
    }
    __syncthreads();

    // S = Q K^T : 16 x 64 (4 col tiles x 4 k-steps)
    v8f sacc[4];
#pragma unroll
    for (int nt = 0; nt < 4; ++nt) sacc[nt] = zero8();
#pragma unroll
    for (int kc = 0; kc < 4; ++kc)
#pragma unroll
      for (int nt = 0; nt < 4; ++nt) {
        v16bf bk = *(const v16bf*)&Ks[(nt * 16 + lr) * 128 + kc * 32 + lh * 16];
        sacc[nt] = wmma_bf16(aq[kc], bk, sacc[nt]);
      }

    // online softmax: each VGPR index i covers rows i (lanes 0-15) / i+8 (16-31)
#pragma unroll
    for (int i = 0; i < 8; ++i) {
      float m = fmaxf(fmaxf(sacc[0][i], sacc[1][i]), fmaxf(sacc[2][i], sacc[3][i]));
#pragma unroll
      for (int off = 1; off < 16; off <<= 1) m = fmaxf(m, __shfl_xor(m, off, 32));
      float mnew = fmaxf(mrow[i], m);
      float corr = __expf(mrow[i] - mnew);
      mrow[i] = mnew;
      float rsum = 0.f;
#pragma unroll
      for (int nt = 0; nt < 4; ++nt) {
        float p = __expf(sacc[nt][i] - mnew);
        sacc[nt][i] = p; rsum += p;
      }
#pragma unroll
      for (int off = 1; off < 16; off <<= 1) rsum += __shfl_xor(rsum, off, 32);
      lrow[i] = lrow[i] * corr + rsum;
#pragma unroll
      for (int n2 = 0; n2 < 8; ++n2) oacc[n2][i] *= corr;
    }

    // relayout P (C layout -> A fragments) via per-wave LDS round trip
    __bf16* pbase = &Ps[wave * 16 * 64];
#pragma unroll
    for (int nt = 0; nt < 4; ++nt)
#pragma unroll
      for (int i = 0; i < 8; ++i)
        pbase[(i + lh * 8) * 64 + nt * 16 + lr] = (__bf16)sacc[nt][i];
    v16bf pa[2];
#pragma unroll
    for (int kc2 = 0; kc2 < 2; ++kc2) {
      const __bf16* pp = pbase + lr * 64 + kc2 * 32;
      pa[kc2] = *(const v16bf*)(pp + lh * 16);
    }

    // O += P @ V : V B-fragments straight from transposed global (L2-resident)
#pragma unroll
    for (int kc2 = 0; kc2 < 2; ++kc2)
#pragma unroll
      for (int n2 = 0; n2 < 8; ++n2) {
        const __bf16* vp =
            vhead + (n2 * 16 + lr) * (long)Lseq + kb * 64 + kc2 * 32 + lh * 16;
        v16bf bv = *(const v16bf*)vp;
        oacc[n2] = wmma_bf16(pa[kc2], bv, oacc[n2]);
      }
  }

#pragma unroll
  for (int n2 = 0; n2 < 8; ++n2)
#pragma unroll
    for (int i = 0; i < 8; ++i) {
      long l = qr0 + i + lh * 8;
      O[(rowbase + l) * (long)Dm + h * Kh + n2 * 16 + lr] =
          (__bf16)(oacc[n2][i] / lrow[i]);
    }
}

// ------------------------------------------------------------- mean+max pool
__global__ __launch_bounds__(256) void pool_mean_max(const __bf16* __restrict__ U,
                                                     __bf16* __restrict__ pooled) {
  const int b = blockIdx.y;
  const int d = blockIdx.x * 256 + threadIdx.x;
  const __bf16* p = U + (long)b * Lseq * Dm + d;
  float s = 0.f, mx = -3.0e38f;
  for (int l = 0; l < Lseq; ++l) {
    float v = (float)p[(long)l * Dm];
    s += v; mx = fmaxf(mx, v);
  }
  pooled[b * Dm + d] = (__bf16)(s * (1.f / Lseq) + mx);
}

// -------------------------------------------------- final MLP: 16x64 = 16x1024 @ W^T
__global__ __launch_bounds__(32) void head_mlp(const __bf16* __restrict__ pooled,
                                               const __bf16* __restrict__ Wm,
                                               const float* __restrict__ bm,
                                               float* __restrict__ out) {
  const int lane = threadIdx.x & 31, lh = lane >> 4, lr = lane & 15;
  v8f acc[4];
#pragma unroll
  for (int i = 0; i < 4; ++i) acc[i] = zero8();
  for (int k0 = 0; k0 < Dm; k0 += 32) {
    const __bf16* ap = pooled + lr * Dm + k0;
    v16bf a = *(const v16bf*)(ap + lh * 16);
#pragma unroll
    for (int fn = 0; fn < 4; ++fn) {
      v16bf bfr = *(const v16bf*)(Wm + (fn * 16 + lr) * Dm + k0 + lh * 16);
      acc[fn] = wmma_bf16(a, bfr, acc[fn]);
    }
  }
#pragma unroll
  for (int fn = 0; fn < 4; ++fn)
#pragma unroll
    for (int i = 0; i < 8; ++i) {
      int row = i + lh * 8, col = fn * 16 + lr;
      out[row * OUTN + col] = acc[fn][i] + bm[col];
    }
}

// ---------------------------------------------------------------------- host
extern "C" void kernel_launch(void* const* d_in, const int* in_sizes, int n_in,
                              void* d_out, int out_size, void* d_ws, size_t ws_size,
                              hipStream_t stream) {
  (void)in_sizes; (void)n_in; (void)out_size; (void)ws_size;
  const float* x0 = (const float*)d_in[0];
  const float* x1 = (const float*)d_in[1];
  const float* x2 = (const float*)d_in[2];
  const float* Wk = (const float*)d_in[5];
  const float* Wq = (const float*)d_in[6];
  const float* Wv = (const float*)d_in[7];
  const float* Wu = (const float*)d_in[8];
  const float* bu = (const float*)d_in[9];
  const float* Wmlp = (const float*)d_in[10];
  const float* bmlp = (const float*)d_in[11];

  char* ws = (char*)d_ws;
  const size_t MB = 1024ull * 1024ull;
  __bf16* Wk_b = (__bf16*)(ws + 0 * MB);
  __bf16* Wq_b = (__bf16*)(ws + 2 * MB);
  __bf16* Wv_b = (__bf16*)(ws + 4 * MB);
  __bf16* Wu_b = (__bf16*)(ws + 6 * MB);
  __bf16* Wm_b = (__bf16*)(ws + 8 * MB);
  __bf16* pooled = (__bf16*)(ws + 9 * MB);
  __bf16* Qb = (__bf16*)(ws + 16 * MB);
  __bf16* Kb = (__bf16*)(ws + 48 * MB);
  __bf16* Vt = (__bf16*)(ws + 80 * MB);   // [b, h, kdim, l]
  __bf16* Ob = (__bf16*)(ws + 112 * MB);
  __bf16* Ub = Qb;  // Q dead after attention -> reuse for unify output

  cvt_f32_bf16<<<1024, 256, 0, stream>>>(Wk, Wk_b, 1024 * 1024);
  cvt_f32_bf16<<<1024, 256, 0, stream>>>(Wq, Wq_b, 1024 * 1024);
  cvt_f32_bf16<<<1024, 256, 0, stream>>>(Wv, Wv_b, 1024 * 1024);
  cvt_f32_bf16<<<1024, 256, 0, stream>>>(Wu, Wu_b, 1024 * 1024);
  cvt_f32_bf16<<<64, 256, 0, stream>>>(Wmlp, Wm_b, 64 * 1024);

  dim3 g(16384 / 128, 1024 / 128);  // M=B*L, N=D
  // q = x2 @ Wk^T (* scale), key = x1 @ Wq^T (* scale), v = x0 @ Wv^T
  gemm_xWt<float, 1, false><<<g, 256, 0, stream>>>(x2, Wk_b, nullptr, QK_SCALE,
                                                   Qb, 1024, 1024);
  gemm_xWt<float, 1, false><<<g, 256, 0, stream>>>(x1, Wq_b, nullptr, QK_SCALE,
                                                   Kb, 1024, 1024);
  gemm_xWt<float, 2, false><<<g, 256, 0, stream>>>(x0, Wv_b, nullptr, 1.0f,
                                                   Vt, 1024, 1024);

  attn_flash<<<dim3(8, Hn, 16), 256, 0, stream>>>(Qb, Kb, Vt, Ob);

  gemm_xWt<__bf16, 1, true><<<g, 256, 0, stream>>>(Ob, Wu_b, bu, 1.0f,
                                                   Ub, 1024, 1024);

  pool_mean_max<<<dim3(4, 16), 256, 0, stream>>>(Ub, pooled);

  head_mlp<<<1, 32, 0, stream>>>(pooled, Wm_b, bmlp, (float*)d_out);
}
